// BitnetMultiAttention_68564857913635
// MI455X (gfx1250) — 8-bit matrix operand order repaired and checked by reading, compiled, not run on hardware
//
#include <hip/hip_runtime.h>
#include <hip/hip_bf16.h>


typedef int v8i __attribute__((ext_vector_type(8)));

#define TOKENS   32768
#define EDIM     1024
#define NHEAD    16
#define HDIM     64
#define EPSF     1e-5f

union AReg { v8i v; int2 d2[4]; };
union BReg { v8i v; int4 d4[2]; };
union CReg { v8i v; int  s[8];  };

__global__ __launch_bounds__(256)
void bn_wabs_partial(const float* __restrict__ w0, const float* __restrict__ w1,
                     const float* __restrict__ w2, const float* __restrict__ w3,
                     float* __restrict__ partial)
{
    __shared__ float red[256];
    const int mat = blockIdx.x >> 6;
    const int chunk = blockIdx.x & 63;
    const float* w = (mat == 0) ? w0 : (mat == 1) ? w1 : (mat == 2) ? w2 : w3;
    const long base = (long)chunk * 16384;
    float s = 0.0f;
    for (int i = threadIdx.x; i < 16384; i += 256) s += fabsf(w[base + i]);
    red[threadIdx.x] = s;
    __syncthreads();
    for (int st = 128; st; st >>= 1) {
        if (threadIdx.x < st) red[threadIdx.x] += red[threadIdx.x + st];
        __syncthreads();
    }
    if (threadIdx.x == 0) partial[blockIdx.x] = red[0];
}

__global__ void bn_wscale(const float* __restrict__ partial, float* __restrict__ wrecip)
{
    if (threadIdx.x < 4) {
        float s = 0.0f;
        for (int i = 0; i < 64; ++i) s += partial[threadIdx.x * 64 + i];
        wrecip[threadIdx.x] = fmaxf(s * (1.0f / 1048576.0f), 1e-5f);
    }
}

__global__ __launch_bounds__(256)
void bn_wquant(const float* __restrict__ w, const float* __restrict__ wrecip, int mat,
               signed char* __restrict__ wq)
{
    const float sc = 1.0f / wrecip[mat];
    const long idx = ((long)blockIdx.x * 256 + threadIdx.x) * 4;
    const float4 v = *(const float4*)(w + idx);
    char4 c;
    c.x = (signed char)fminf(fmaxf(rintf(v.x * sc), -1.0f), 1.0f);
    c.y = (signed char)fminf(fmaxf(rintf(v.y * sc), -1.0f), 1.0f);
    c.z = (signed char)fminf(fmaxf(rintf(v.z * sc), -1.0f), 1.0f);
    c.w = (signed char)fminf(fmaxf(rintf(v.w * sc), -1.0f), 1.0f);
    *(char4*)(wq + idx) = c;
}

__global__ __launch_bounds__(256)
void bn_actquant(const float* __restrict__ x, signed char* __restrict__ xq,
                 float* __restrict__ recip)
{
    const int lane = threadIdx.x & 31;
    const int wave = threadIdx.x >> 5;
    const long t = (long)blockIdx.x * 8 + wave;
    const float* xr = x + t * EDIM;

    float4 v[8];
    float amax = 0.0f;
#pragma unroll
    for (int i = 0; i < 8; ++i) {
        v[i] = *(const float4*)(xr + (i * 32 + lane) * 4);
        amax = fmaxf(amax, fmaxf(fmaxf(fabsf(v[i].x), fabsf(v[i].y)),
                                 fmaxf(fabsf(v[i].z), fabsf(v[i].w))));
    }
#pragma unroll
    for (int m = 16; m; m >>= 1) amax = fmaxf(amax, __shfl_xor(amax, m, 32));
    const float am = fmaxf(amax, 1e-5f);
    const float sc = 127.0f / am;

    signed char* o = xq + t * EDIM;
#pragma unroll
    for (int i = 0; i < 8; ++i) {
        char4 c;
        c.x = (signed char)fminf(fmaxf(rintf(v[i].x * sc), -128.0f), 127.0f);
        c.y = (signed char)fminf(fmaxf(rintf(v[i].y * sc), -128.0f), 127.0f);
        c.z = (signed char)fminf(fmaxf(rintf(v[i].z * sc), -128.0f), 127.0f);
        c.w = (signed char)fminf(fmaxf(rintf(v[i].w * sc), -128.0f), 127.0f);
        *(char4*)(o + (i * 32 + lane) * 4) = c;
    }
    if (lane == 0) recip[t] = am * (1.0f / 127.0f);
}

__global__ __launch_bounds__(256)
void bn_bitgemm_i8(const signed char* __restrict__ A,
                   const float* __restrict__ recipX,
                   const signed char* __restrict__ W,
                   const float* __restrict__ wrecip,
                   const float* __restrict__ bias,
                   float* __restrict__ out)
{
    const int K = EDIM, O = EDIM;
    const int lane = threadIdx.x & 31;
    const int wave = threadIdx.x >> 5;
    const int half = lane >> 4;
    const int l16  = lane & 15;
    const long tokBase = ((long)blockIdx.x * 8 + wave) * 16;
    const int  nBase   = blockIdx.y * 64;

    CReg acc[4];
#pragma unroll
    for (int t = 0; t < 4; ++t)
        acc[t].v = (v8i){0, 0, 0, 0, 0, 0, 0, 0};

    const signed char* aPtr = A + (tokBase + l16) * (long)K + half * 16;
    const signed char* bRow[4];
#pragma unroll
    for (int t = 0; t < 4; ++t)
        bRow[t] = W + (long)(nBase + t * 16 + l16) * K + half * 16;

    AReg a[2];
    BReg b[2][4];

#pragma unroll
    for (int s = 0; s < 4; ++s) a[0].d2[s] = *(const int2*)(aPtr + (s & 1) * 8 + (s >> 1) * 32);
#pragma unroll
    for (int t = 0; t < 4; ++t) {
        b[0][t].d4[0] = *(const int4*)(bRow[t]);
        b[0][t].d4[1] = *(const int4*)(bRow[t] + 32);
    }

#pragma unroll
    for (int ks = 0; ks < 16; ++ks) {
        const int cur = ks & 1;
        const int nxt = cur ^ 1;
        if (ks < 15) {
            const int k0n = (ks + 1) * 64;
#pragma unroll
            for (int s = 0; s < 4; ++s)
                a[nxt].d2[s] = *(const int2*)(aPtr + k0n + (s & 1) * 8 + (s >> 1) * 32);
#pragma unroll
            for (int t = 0; t < 4; ++t) {
                b[nxt][t].d4[0] = *(const int4*)(bRow[t] + k0n);
                b[nxt][t].d4[1] = *(const int4*)(bRow[t] + k0n + 32);
            }
        }
#pragma unroll
        for (int t = 0; t < 4; ++t)
            acc[t].v = __builtin_amdgcn_wmma_i32_16x16x64_iu8(
                true, a[cur].v, true, b[cur][t].v, acc[t].v, false, false);
    }

    const float wr = wrecip[0];
    float rx[8];
#pragma unroll
    for (int j = 0; j < 8; ++j) rx[j] = recipX[tokBase + j + half * 8];

#pragma unroll
    for (int t = 0; t < 4; ++t) {
        const int n = nBase + t * 16 + l16;
        const float bn = bias[n];
#pragma unroll
        for (int j = 0; j < 8; ++j) {
            const long row = tokBase + j + half * 8;
            out[row * (long)O + n] = (float)acc[t].s[j] * rx[j] * wr + bn;
        }
    }
}

__global__ __launch_bounds__(256)
void bn_attn_ln_quant(const float* __restrict__ q, const float* __restrict__ k,
                      const float* __restrict__ v, const float* __restrict__ gamma,
                      const float* __restrict__ beta,
                      signed char* __restrict__ xq, float* __restrict__ recip)
{
    __shared__ float sq[NHEAD * 68], sk[NHEAD * 68], sv[NHEAD * 68];
    __shared__ float sattn[256];
    __shared__ float red[256], red2[256];

    const int tid = threadIdx.x;
    const long t = blockIdx.x;
    const float* qr = q + t * EDIM;
    const float* kr = k + t * EDIM;
    const float* vr = v + t * EDIM;

#pragma unroll
    for (int i = 0; i < 4; ++i) {
        const int idx = tid + i * 256;
        const int hh = idx >> 6, dd = idx & 63;
        sq[hh * 68 + dd] = qr[idx];
        sk[hh * 68 + dd] = kr[idx];
        sv[hh * 68 + dd] = vr[idx];
    }
    __syncthreads();

    const int h = tid >> 4, g = tid & 15;
    const float* qh = sq + h * 68;
    const float* kg = sk + g * 68;
    float s = 0.0f;
#pragma unroll 8
    for (int d = 0; d < 64; ++d) s += qh[d] * kg[d];
    s *= 0.125f;

    float mx = s;
#pragma unroll
    for (int m = 8; m; m >>= 1) mx = fmaxf(mx, __shfl_xor(mx, m, 16));
    const float e = expf(s - mx);
    float sum = e;
#pragma unroll
    for (int m = 8; m; m >>= 1) sum += __shfl_xor(sum, m, 16);
    sattn[tid] = e / sum;
    __syncthreads();

    float xv[4], lsum = 0.0f, lsq = 0.0f;
#pragma unroll
    for (int j = 0; j < 4; ++j) {
        const int idx = tid * 4 + j;
        const int hh = idx >> 6, dd = idx & 63;
        const float* arow = sattn + hh * 16;
        float acc = 0.0f;
#pragma unroll
        for (int gg = 0; gg < 16; ++gg) acc += arow[gg] * sv[gg * 68 + dd];
        xv[j] = acc;
        lsum += acc;
        lsq += acc * acc;
    }
    red[tid] = lsum; red2[tid] = lsq;
    __syncthreads();
    for (int st = 128; st; st >>= 1) {
        if (tid < st) { red[tid] += red[tid + st]; red2[tid] += red2[tid + st]; }
        __syncthreads();
    }
    const float mu   = red[0] * (1.0f / 1024.0f);
    const float var  = red2[0] * (1.0f / 1024.0f) - mu * mu;
    const float rstd = rsqrtf(var + EPSF);

    float y[4], amax = 0.0f;
#pragma unroll
    for (int j = 0; j < 4; ++j) {
        const int idx = tid * 4 + j;
        y[j] = (xv[j] - mu) * rstd * gamma[idx] + beta[idx];
        amax = fmaxf(amax, fabsf(y[j]));
    }
    __syncthreads();
    red[tid] = amax;
    __syncthreads();
    for (int st = 128; st; st >>= 1) {
        if (tid < st) red[tid] = fmaxf(red[tid], red[tid + st]);
        __syncthreads();
    }
    const float am = fmaxf(red[0], 1e-5f);
    const float sc = 127.0f / am;
    char4 c;
    c.x = (signed char)fminf(fmaxf(rintf(y[0] * sc), -128.0f), 127.0f);
    c.y = (signed char)fminf(fmaxf(rintf(y[1] * sc), -128.0f), 127.0f);
    c.z = (signed char)fminf(fmaxf(rintf(y[2] * sc), -128.0f), 127.0f);
    c.w = (signed char)fminf(fmaxf(rintf(y[3] * sc), -128.0f), 127.0f);
    *(char4*)(xq + t * EDIM + tid * 4) = c;
    if (tid == 0) recip[t] = am * (1.0f / 127.0f);
}

extern "C" void kernel_launch(void* const* d_in, const int* in_sizes, int n_in,
                              void* d_out, int out_size, void* d_ws, size_t ws_size,
                              hipStream_t stream)
{
    (void)in_sizes; (void)n_in; (void)out_size; (void)ws_size;

    const float* query = (const float*)d_in[0];
    const float* key   = (const float*)d_in[1];
    const float* value = (const float*)d_in[2];
    const float* Wq    = (const float*)d_in[3];
    const float* bq    = (const float*)d_in[4];
    const float* Wk    = (const float*)d_in[5];
    const float* bk    = (const float*)d_in[6];
    const float* Wv    = (const float*)d_in[7];
    const float* bv    = (const float*)d_in[8];
    const float* gamma = (const float*)d_in[9];
    const float* beta  = (const float*)d_in[10];
    const float* Wo    = (const float*)d_in[11];
    const float* bo    = (const float*)d_in[12];
    float* out = (float*)d_out;

    char* ws = (char*)d_ws;
    size_t off = 0;
    auto take = [&](size_t sz) -> char* {
        char* p = ws + off;
        off += (sz + 255) & ~(size_t)255;
        return p;
    };
    const size_t WSZ   = (size_t)EDIM * EDIM;
    const size_t ACT8  = (size_t)TOKENS * EDIM;
    const size_t ACT32 = (size_t)TOKENS * EDIM * 4;
    float* wpartial       = (float*)take(4 * 64 * sizeof(float));
    float* wrecip         = (float*)take(4 * sizeof(float));
    signed char* Wq8      = (signed char*)take(WSZ);
    signed char* Wk8      = (signed char*)take(WSZ);
    signed char* Wv8      = (signed char*)take(WSZ);
    signed char* Wo8      = (signed char*)take(WSZ);
    signed char* xq_q     = (signed char*)take(ACT8);
    signed char* xq_k     = (signed char*)take(ACT8);
    signed char* xq_v     = (signed char*)take(ACT8);
    float* rq             = (float*)take(TOKENS * sizeof(float));
    float* rk             = (float*)take(TOKENS * sizeof(float));
    float* rv             = (float*)take(TOKENS * sizeof(float));
    float* qf             = (float*)take(ACT32);
    float* kf             = (float*)take(ACT32);
    float* vf             = (float*)take(ACT32);
    signed char* x8       = (signed char*)take(ACT8);
    float* rx             = (float*)take(TOKENS * sizeof(float));

    bn_wabs_partial<<<dim3(4 * 64), 256, 0, stream>>>(Wq, Wk, Wv, Wo, wpartial);
    bn_wscale<<<1, 32, 0, stream>>>(wpartial, wrecip);

    bn_wquant<<<dim3(1024), 256, 0, stream>>>(Wq, wrecip, 0, Wq8);
    bn_wquant<<<dim3(1024), 256, 0, stream>>>(Wk, wrecip, 1, Wk8);
    bn_wquant<<<dim3(1024), 256, 0, stream>>>(Wv, wrecip, 2, Wv8);
    bn_wquant<<<dim3(1024), 256, 0, stream>>>(Wo, wrecip, 3, Wo8);

    bn_actquant<<<dim3(TOKENS / 8), 256, 0, stream>>>(query, xq_q, rq);
    bn_actquant<<<dim3(TOKENS / 8), 256, 0, stream>>>(key,   xq_k, rk);
    bn_actquant<<<dim3(TOKENS / 8), 256, 0, stream>>>(value, xq_v, rv);

    const dim3 gemmGrid(TOKENS / (16 * 8), EDIM / 64);
    bn_bitgemm_i8<<<gemmGrid, 256, 0, stream>>>(xq_q, rq, Wq8, wrecip + 0, bq, qf);
    bn_bitgemm_i8<<<gemmGrid, 256, 0, stream>>>(xq_k, rk, Wk8, wrecip + 1, bk, kf);
    bn_bitgemm_i8<<<gemmGrid, 256, 0, stream>>>(xq_v, rv, Wv8, wrecip + 2, bv, vf);

    bn_attn_ln_quant<<<dim3(TOKENS), 256, 0, stream>>>(qf, kf, vf, gamma, beta, x8, rx);

    bn_bitgemm_i8<<<gemmGrid, 256, 0, stream>>>(x8, rx, Wo8, wrecip + 3, bo, out);
}
